// EarthSpecificLayer_63007170232402
// MI455X (gfx1250) — compile-verified, matrix operand order checked by reading, not run on hardware
//
#include <hip/hip_runtime.h>
#include <hip/hip_bf16.h>

typedef __attribute__((ext_vector_type(16))) _Float16 v16h;
typedef __attribute__((ext_vector_type(8)))  _Float16 v8h;
typedef __attribute__((ext_vector_type(8)))  float    v8f;

#define N_WIN  960
#define N_TOK  144
#define CDIM   192
#define HDIM   32
#define NLON   15
#define L_TOK  131040   // 8*91*180

// ---------------- helpers ----------------

__device__ __forceinline__ v8f zero8() {
  v8f z = {0.f,0.f,0.f,0.f,0.f,0.f,0.f,0.f};
  return z;
}

__device__ __forceinline__ v16h zero16h() {
  v16h r;
#pragma unroll
  for (int i = 0; i < 16; ++i) r[i] = (_Float16)0.f;
  return r;
}

__device__ __forceinline__ v8f wmma32(v16h a, v16h b, v8f c) {
  // D = A(16x32 f16) * B(32x16 f16) + C(16x16 f32)
  return __builtin_amdgcn_wmma_f32_16x16x32_f16(false, a, false, b, (short)0, c, false, false);
}

// 16-bit A/B fragment: per lane two contiguous 8xf16 chunks at +0 and +16 elems
__device__ __forceinline__ v16h load_ab(const _Float16* p) {
  v8h lo = *(const v8h*)(p);
  v8h hi = *(const v8h*)(p + 16);
  v16h r;
#pragma unroll
  for (int i = 0; i < 8; ++i) { r[i] = lo[i]; r[i + 8] = hi[i]; }
  return r;
}

// A fragment gathered+converted from an fp32 row (must be non-null)
__device__ __forceinline__ v16h load_a_f32(const float* q) {
  v16h r;
#pragma unroll
  for (int i = 0; i < 8; ++i) { r[i] = (_Float16)q[i]; r[i + 8] = (_Float16)q[i + 16]; }
  return r;
}

// token t in window (zw,hw,lw) -> flat row of x, or -1 if in lat padding
__device__ __forceinline__ int tok_row(int zw, int hw, int lw, int t) {
  int zi = t / 72, hi = (t / 12) % 6, wi = t % 12;
  int z   = zw * 2 + zi;
  int lat = hw * 6 + hi - 2;     // PAD_TOP = 2
  int w   = lw * 12 + wi;
  if (lat < 0 || lat >= 91) return -1;
  return (z * 91 + lat) * 180 + w;
}

// ---------------- kernel 0: fp32 -> f16 weight conversion ----------------

__global__ void cvt_f32_to_f16(const float* __restrict__ in, _Float16* __restrict__ out, int n) {
  int i = blockIdx.x * blockDim.x + threadIdx.x;
  if (i < n) out[i] = (_Float16)in[i];
}

// ---------------- kernel 1: fused window attention ----------------
// grid (960, 6) = (window, head); block 128 threads = 4 waves

__global__ void __launch_bounds__(128)
attn_kernel(const float* __restrict__ x,
            const _Float16* __restrict__ qkvw16,
            const float* __restrict__ qkv_b,
            const float* __restrict__ bias_table,
            _Float16* __restrict__ attn16)
{
  __shared__ __align__(16) _Float16 sQ[N_TOK * HDIM];     //  9216 B
  __shared__ __align__(16) _Float16 sK[N_TOK * HDIM];     //  9216 B
  __shared__ __align__(16) _Float16 sVt[HDIM * 160];      // 10240 B (V transposed, K padded 144->160)
  __shared__ __align__(16) _Float16 sAtt[4][16 * 160];    // 20480 B (per-wave softmaxed att tile)

  const int lane = threadIdx.x & 31;
  const int wv   = threadIdx.x >> 5;
  const int ln   = lane & 15;
  const int hf   = lane >> 4;
  const int wId  = blockIdx.x;
  const int head = blockIdx.y;
  const int zw   = wId / (16 * NLON);
  const int hw   = (wId / NLON) % 16;
  const int lw   = wId % NLON;
  const int wb   = wId / NLON;             // bias window index in [0,64)

  // zero sVt padding columns 144..159
  for (int i = threadIdx.x; i < 32 * 16; i += blockDim.x)
    sVt[(i >> 4) * 160 + 144 + (i & 15)] = (_Float16)0.f;

  // ---- phase 1: Q,K,V = X @ Wqkv_head^T (+bias, q scaled) ----
  // Per M-strip: gather+convert the 16x192 A strip once, reuse for 6 output tiles.
  for (int mt = wv; mt < 9; mt += 4) {
    const int tq = mt * 16 + ln;
    const int xr = tok_row(zw, hw, lw, tq);
    v16h afr[6];
    if (xr >= 0) {
      const float* arow = x + (size_t)xr * CDIM;
#pragma unroll
      for (int kk = 0; kk < 6; ++kk) afr[kk] = load_a_f32(arow + kk * 32 + hf * 8);
    } else {
#pragma unroll
      for (int kk = 0; kk < 6; ++kk) afr[kk] = zero16h();
    }
#pragma unroll
    for (int st = 0; st < 6; ++st) {         // sel(3) x ntile(2)
      const int sel = st >> 1;
      const int nt  = st & 1;
      const int wrow = sel * CDIM + head * HDIM + nt * 16 + ln;
      const _Float16* brow = qkvw16 + (size_t)wrow * CDIM;
      v16h bfr[6];
#pragma unroll
      for (int kk = 0; kk < 6; ++kk) bfr[kk] = load_ab(brow + kk * 32 + hf * 8);
      v8f c = zero8();
#pragma unroll
      for (int kk = 0; kk < 6; ++kk) c = wmma32(afr[kk], bfr[kk], c);
      const float bias = qkv_b[wrow];
      const float scl  = (sel == 0) ? 0.17677669529663687f : 1.0f;  // hd^-0.5
#pragma unroll
      for (int r = 0; r < 8; ++r) {
        const int m = mt * 16 + r + hf * 8;
        const int n = nt * 16 + ln;
        const float v = (c[r] + bias) * scl;
        if (sel == 0)      sQ[m * HDIM + n] = (_Float16)v;
        else if (sel == 1) sK[m * HDIM + n] = (_Float16)v;
        else               sVt[n * 160 + m] = (_Float16)v;   // transposed V
      }
    }
  }
  __syncthreads();

  // ---- phase 2: per M-tile: att = qK^T + bias, softmax, out = att @ V ----
  for (int mt = wv; mt < 9; mt += 4) {
    v16h aq = load_ab(sQ + (mt * 16 + ln) * HDIM + hf * 8);
    v16h bq[9];
#pragma unroll
    for (int nt = 0; nt < 9; ++nt)
      bq[nt] = load_ab(sK + (nt * 16 + ln) * HDIM + hf * 8);  // B[k][n] = K[n][k]
    v8f att[9];
#pragma unroll
    for (int nt = 0; nt < 9; ++nt) att[nt] = wmma32(aq, bq[nt], zero8());

    // earth-specific position bias (computed analytically)
#pragma unroll
    for (int nt = 0; nt < 9; ++nt) {
      const int tk = nt * 16 + ln;
      const int zj = tk / 72, hj = (tk / 12) % 6, wj = tk % 12;
#pragma unroll
      for (int r = 0; r < 8; ++r) {
        const int tq = mt * 16 + r + hf * 8;
        const int zi = tq / 72, hi = (tq / 12) % 6, wi = tq % 12;
        const int idx = (zi + 2 * zj) * 828 + (hi + 6 * hj) * 23 + (wi - wj + 11);
        att[nt][r] += bias_table[(size_t)idx * 384 + wb * 6 + head];
      }
    }
    // register-resident softmax; row r lives in lanes of one 16-lane half
#pragma unroll
    for (int r = 0; r < 8; ++r) {
      float mx = -3.0e38f;
#pragma unroll
      for (int nt = 0; nt < 9; ++nt) mx = fmaxf(mx, att[nt][r]);
#pragma unroll
      for (int off = 1; off < 16; off <<= 1) mx = fmaxf(mx, __shfl_xor(mx, off));
      float s = 0.f;
#pragma unroll
      for (int nt = 0; nt < 9; ++nt) { float e = __expf(att[nt][r] - mx); att[nt][r] = e; s += e; }
#pragma unroll
      for (int off = 1; off < 16; off <<= 1) s += __shfl_xor(s, off);
      const float inv = 1.f / s;
      const int rl = r + hf * 8;
#pragma unroll
      for (int nt = 0; nt < 9; ++nt)
        sAtt[wv][rl * 160 + nt * 16 + ln] = (_Float16)(att[nt][r] * inv);
      sAtt[wv][rl * 160 + 144 + ln] = (_Float16)0.f;   // zero K padding 144..159
    }
    asm volatile("s_wait_dscnt 0" ::: "memory");       // LDS write->read, same wave
    __builtin_amdgcn_wave_barrier();

    v8f o0 = zero8(), o1 = zero8();
#pragma unroll
    for (int kk = 0; kk < 5; ++kk) {                   // K = 160
      v16h aa = load_ab(&sAtt[wv][ln * 160 + kk * 32 + hf * 8]);
      v16h b0 = load_ab(sVt + ln * 160 + kk * 32 + hf * 8);          // B[k][n]=Vt[n][k]
      v16h b1 = load_ab(sVt + (16 + ln) * 160 + kk * 32 + hf * 8);
      o0 = wmma32(aa, b0, o0);
      o1 = wmma32(aa, b1, o1);
    }
#pragma unroll
    for (int r = 0; r < 8; ++r) {
      const int m = mt * 16 + r + hf * 8;
      _Float16* dst = attn16 + ((size_t)wId * N_TOK + m) * CDIM + head * HDIM;
      dst[ln]      = (_Float16)o0[r];
      dst[16 + ln] = (_Float16)o1[r];
    }
  }
}

// ---------------- kernel 2: proj + residual + layernorm1 -> x1 ----------------
// grid 960 (windows), block 128 (4 waves); wave owns a full 16-row M-tile

__global__ void __launch_bounds__(128)
proj_kernel(const _Float16* __restrict__ attn16,
            const _Float16* __restrict__ projw16,
            const float* __restrict__ proj_b,
            const float* __restrict__ x,
            const float* __restrict__ n1w,
            const float* __restrict__ n1b,
            float* __restrict__ out,        // receives x1 (f32)
            _Float16* __restrict__ x1h)     // x1 (f16) for the MLP GEMM
{
  const int lane = threadIdx.x & 31;
  const int wv   = threadIdx.x >> 5;
  const int ln   = lane & 15;
  const int hf   = lane >> 4;
  const int wId  = blockIdx.x;
  const int zw   = wId / (16 * NLON);
  const int hw   = (wId / NLON) % 16;
  const int lw   = wId % NLON;

  for (int mt = wv; mt < 9; mt += 4) {
    v8f acc[12];
#pragma unroll
    for (int i = 0; i < 12; ++i) acc[i] = zero8();
    const _Float16* arow = attn16 + ((size_t)wId * N_TOK + mt * 16 + ln) * CDIM;
#pragma unroll
    for (int kk = 0; kk < 6; ++kk) {
      v16h a = load_ab(arow + kk * 32 + hf * 8);
      __builtin_prefetch(arow + (kk + 1) * 32, 0, 1);
#pragma unroll
      for (int g = 0; g < 2; ++g) {        // batch B frags in groups of 6
        v16h bf[6];
#pragma unroll
        for (int j = 0; j < 6; ++j)
          bf[j] = load_ab(projw16 + (size_t)((g * 6 + j) * 16 + ln) * CDIM + kk * 32 + hf * 8);
#pragma unroll
        for (int j = 0; j < 6; ++j)
          acc[g * 6 + j] = wmma32(a, bf[j], acc[g * 6 + j]);
      }
    }
#pragma unroll
    for (int nt = 0; nt < 12; ++nt) {
      const float pb = proj_b[nt * 16 + ln];
#pragma unroll
      for (int r = 0; r < 8; ++r) acc[nt][r] += pb;
    }
#pragma unroll
    for (int r = 0; r < 8; ++r) {
      const int tq = mt * 16 + r + hf * 8;
      const int xr = tok_row(zw, hw, lw, tq);
      if (xr < 0) continue;                  // padded row: cropped, no output
      float s = 0.f;
#pragma unroll
      for (int nt = 0; nt < 12; ++nt) s += acc[nt][r];
#pragma unroll
      for (int off = 1; off < 16; off <<= 1) s += __shfl_xor(s, off);
      const float mu = s * (1.f / 192.f);
      float v2 = 0.f;
#pragma unroll
      for (int nt = 0; nt < 12; ++nt) { float d = acc[nt][r] - mu; v2 += d * d; }
#pragma unroll
      for (int off = 1; off < 16; off <<= 1) v2 += __shfl_xor(v2, off);
      const float rn = rsqrtf(v2 * (1.f / 192.f) + 1e-5f);
#pragma unroll
      for (int nt = 0; nt < 12; ++nt) {
        const int n = nt * 16 + ln;
        const float lnv = (acc[nt][r] - mu) * rn * n1w[n] + n1b[n];
        const float x1  = x[(size_t)xr * CDIM + n] + lnv;
        out[(size_t)xr * CDIM + n] = x1;
        x1h[(size_t)xr * CDIM + n] = (_Float16)x1;
      }
    }
  }
}

// ---------------- kernel 3: fused MLP + layernorm2 + residual ----------------
// grid 4095, block 64 (2 waves); wave owns one 16-row M-tile; h1 stays in LDS

__global__ void __launch_bounds__(64)
mlp_kernel(const _Float16* __restrict__ x1h,
           const _Float16* __restrict__ w1h,
           const float* __restrict__ b1,
           const _Float16* __restrict__ w2h,
           const float* __restrict__ b2,
           const float* __restrict__ n2w,
           const float* __restrict__ n2b,
           float* __restrict__ out)         // holds x1; updated in place
{
  __shared__ __align__(16) _Float16 sH[2][16 * 768];   // 2 x 24576 B
  const int lane = threadIdx.x & 31;
  const int wv   = threadIdx.x >> 5;
  const int ln   = lane & 15;
  const int hf   = lane >> 4;
  const int tile = blockIdx.x * 2 + wv;                // < 8190 = 131040/16
  const size_t row0 = (size_t)tile * 16;

  // phase 1: h1 = gelu(x1 @ W1^T + b1), staged in LDS as f16.
  // A strip (16x192 f16) loaded once, reused for all 48 N tiles.
  const _Float16* arow = x1h + (row0 + ln) * CDIM;
  v16h afr[6];
#pragma unroll
  for (int kk = 0; kk < 6; ++kk) afr[kk] = load_ab(arow + kk * 32 + hf * 8);

  for (int nt = 0; nt < 48; ++nt) {
    const _Float16* brow = w1h + (size_t)(nt * 16 + ln) * CDIM;
    v16h bfr[6];
#pragma unroll
    for (int kk = 0; kk < 6; ++kk) bfr[kk] = load_ab(brow + kk * 32 + hf * 8);
    v8f c = zero8();
#pragma unroll
    for (int kk = 0; kk < 6; ++kk) c = wmma32(afr[kk], bfr[kk], c);
    const float bb = b1[nt * 16 + ln];
#pragma unroll
    for (int r = 0; r < 8; ++r) {
      const float v = c[r] + bb;
      const float g = 0.5f * v * (1.f + erff(v * 0.70710678118654752f));  // exact gelu
      sH[wv][(r + hf * 8) * 768 + nt * 16 + ln] = (_Float16)g;
    }
  }
  asm volatile("s_wait_dscnt 0" ::: "memory");
  __builtin_amdgcn_wave_barrier();

  // phase 2: mlp_out = h1 @ W2^T + b2 ; out = x1 + LN2(mlp_out)
  v8f acc[12];
#pragma unroll
  for (int i = 0; i < 12; ++i) acc[i] = zero8();
  for (int kk = 0; kk < 24; ++kk) {
    v16h a = load_ab(&sH[wv][ln * 768 + kk * 32 + hf * 8]);
#pragma unroll
    for (int g = 0; g < 2; ++g) {          // batch B frags in groups of 6
      v16h bf[6];
#pragma unroll
      for (int j = 0; j < 6; ++j)
        bf[j] = load_ab(w2h + (size_t)((g * 6 + j) * 16 + ln) * 768 + kk * 32 + hf * 8);
#pragma unroll
      for (int j = 0; j < 6; ++j)
        acc[g * 6 + j] = wmma32(a, bf[j], acc[g * 6 + j]);
    }
  }
#pragma unroll
  for (int nt = 0; nt < 12; ++nt) {
    const float pb = b2[nt * 16 + ln];
#pragma unroll
    for (int r = 0; r < 8; ++r) acc[nt][r] += pb;
  }
#pragma unroll
  for (int r = 0; r < 8; ++r) {
    float s = 0.f;
#pragma unroll
    for (int nt = 0; nt < 12; ++nt) s += acc[nt][r];
#pragma unroll
    for (int off = 1; off < 16; off <<= 1) s += __shfl_xor(s, off);
    const float mu = s * (1.f / 192.f);
    float v2 = 0.f;
#pragma unroll
    for (int nt = 0; nt < 12; ++nt) { float d = acc[nt][r] - mu; v2 += d * d; }
#pragma unroll
    for (int off = 1; off < 16; off <<= 1) v2 += __shfl_xor(v2, off);
    const float rn = rsqrtf(v2 * (1.f / 192.f) + 1e-5f);
    const size_t row = row0 + r + hf * 8;
#pragma unroll
    for (int nt = 0; nt < 12; ++nt) {
      const int n = nt * 16 + ln;
      const float ln2 = (acc[nt][r] - mu) * rn * n2w[n] + n2b[n];
      const size_t idx = row * CDIM + n;
      out[idx] = out[idx] + ln2;           // x1 + LN2(mlp)
    }
  }
}

// ---------------- host launcher ----------------

extern "C" void kernel_launch(void* const* d_in, const int* in_sizes, int n_in,
                              void* d_out, int out_size, void* d_ws, size_t ws_size,
                              hipStream_t stream) {
  (void)in_sizes; (void)n_in; (void)out_size; (void)ws_size;
  const float* x          = (const float*)d_in[0];
  const float* qkv_w      = (const float*)d_in[1];
  const float* qkv_b      = (const float*)d_in[2];
  const float* proj_w     = (const float*)d_in[3];
  const float* proj_b     = (const float*)d_in[4];
  const float* bias_table = (const float*)d_in[5];
  const float* n1w        = (const float*)d_in[6];
  const float* n1b        = (const float*)d_in[7];
  const float* n2w        = (const float*)d_in[8];
  const float* n2b        = (const float*)d_in[9];
  const float* mlp_w1     = (const float*)d_in[10];
  const float* mlp_b1     = (const float*)d_in[11];
  const float* mlp_w2     = (const float*)d_in[12];
  const float* mlp_b2     = (const float*)d_in[13];
  float* out = (float*)d_out;

  // workspace layout (f16 elements)
  _Float16* wsf     = (_Float16*)d_ws;
  _Float16* qkvw16  = wsf;                              // 576*192
  _Float16* projw16 = qkvw16 + 576 * 192;               // 192*192
  _Float16* w1h     = projw16 + 192 * 192;              // 768*192
  _Float16* w2h     = w1h + 768 * 192;                  // 192*768
  _Float16* attn16  = w2h + 192 * 768;                  // 960*144*192
  _Float16* x1h     = attn16 + (size_t)N_WIN * N_TOK * CDIM;  // 131040*192

  cvt_f32_to_f16<<<(576 * 192 + 255) / 256, 256, 0, stream>>>(qkv_w,  qkvw16,  576 * 192);
  cvt_f32_to_f16<<<(192 * 192 + 255) / 256, 256, 0, stream>>>(proj_w, projw16, 192 * 192);
  cvt_f32_to_f16<<<(768 * 192 + 255) / 256, 256, 0, stream>>>(mlp_w1, w1h,     768 * 192);
  cvt_f32_to_f16<<<(192 * 768 + 255) / 256, 256, 0, stream>>>(mlp_w2, w2h,     192 * 768);

  attn_kernel<<<dim3(N_WIN, 6), 128, 0, stream>>>(x, qkvw16, qkv_b, bias_table, attn16);
  proj_kernel<<<N_WIN, 128, 0, stream>>>(attn16, projw16, proj_b, x, n1w, n1b, out, x1h);
  mlp_kernel<<<L_TOK / 32, 64, 0, stream>>>(x1h, w1h, mlp_b1, w2h, mlp_b2, n2w, n2b, out);
}
